// Attention_2413771621092
// MI455X (gfx1250) — matrix operand order repaired and checked by reading, not compiled, not run on hardware
//
#include <hip/hip_runtime.h>
#include <hip/hip_bf16.h>

typedef __attribute__((ext_vector_type(8)))  __bf16 v8bf;
typedef __attribute__((ext_vector_type(16))) __bf16 v16bf;
typedef __attribute__((ext_vector_type(8)))  float  v8f;
typedef __attribute__((ext_vector_type(4)))  int    v4i;

#define BB   2
#define SS   2048
#define DD   1024
#define HH   16
#define HDD  64
#define ROWS (BB * SS)   // 4096
#define NEGV (-1e20f)
#define SCALE 0.125f     // 64^-0.5
#define EPSV  1e-5f

static __device__ __forceinline__ v16bf join8(v8bf lo, v8bf hi) {
    return __builtin_shufflevector(lo, hi, 0,1,2,3,4,5,6,7,8,9,10,11,12,13,14,15);
}
static __device__ __forceinline__ v8f vzero8() {
    v8f z = {0.f,0.f,0.f,0.f,0.f,0.f,0.f,0.f};
    return z;
}

// ---------------------------------------------------------------------------
// Async global->LDS staging (CDNA5 GLOBAL_LOAD_ASYNC_TO_LDS_B128, ASYNCcnt).
// Builtin signature (from clang diagnostic): (v4i AS(1)*, v4i AS(3)*, Ii, Ii).
// ---------------------------------------------------------------------------
#if defined(__AMDGCN__) && __has_builtin(__builtin_amdgcn_global_load_async_to_lds_b128)
#define GAS __attribute__((address_space(1)))
#define LAS __attribute__((address_space(3)))
static __device__ __forceinline__ void cp16_to_lds(const __bf16* g, __bf16* l) {
    __builtin_amdgcn_global_load_async_to_lds_b128((GAS v4i*)g, (LAS v4i*)l, 0, 0);
}
static __device__ __forceinline__ void wait_stage() {
#if __has_builtin(__builtin_amdgcn_s_wait_asynccnt)
    __builtin_amdgcn_s_wait_asynccnt(0);
#else
    asm volatile("s_wait_asynccnt 0" ::: "memory");
#endif
}
#else
static __device__ __forceinline__ void cp16_to_lds(const __bf16* g, __bf16* l) {
    *(v8bf*)l = *(const v8bf*)g;
}
static __device__ __forceinline__ void wait_stage() {}
#endif

static __device__ __forceinline__ void wave_lds_fence() {
    asm volatile("s_wait_dscnt 0" ::: "memory");
}

// ---------------------------------------------------------------------------
// LayerNorm: one 256-thread block per row of [4096, 1024], f32 -> bf16
// ---------------------------------------------------------------------------
__global__ __launch_bounds__(256) void ln_bf16(const float* __restrict__ x,
                                               const float* __restrict__ gamma,
                                               const float* __restrict__ beta,
                                               __bf16* __restrict__ y) {
    __shared__ float red[8];
    __shared__ float red2[8];
    const int row = blockIdx.x;
    const int t   = threadIdx.x;
    const float4 v = *(const float4*)(x + (size_t)row * DD + t * 4);

    float s = v.x + v.y + v.z + v.w;
    #pragma unroll
    for (int o = 16; o >= 1; o >>= 1) s += __shfl_xor(s, o);
    if ((t & 31) == 0) red[t >> 5] = s;
    __syncthreads();
    float tot = 0.f;
    #pragma unroll
    for (int i = 0; i < 8; ++i) tot += red[i];
    const float mu = tot * (1.0f / DD);

    const float dx = v.x - mu, dy = v.y - mu, dz = v.z - mu, dw = v.w - mu;
    float ss = dx*dx + dy*dy + dz*dz + dw*dw;
    #pragma unroll
    for (int o = 16; o >= 1; o >>= 1) ss += __shfl_xor(ss, o);
    if ((t & 31) == 0) red2[t >> 5] = ss;
    __syncthreads();
    float tot2 = 0.f;
    #pragma unroll
    for (int i = 0; i < 8; ++i) tot2 += red2[i];
    const float rstd = rsqrtf(tot2 * (1.0f / DD) + EPSV);

    const int c = t * 4;
    float4 g  = *(const float4*)(gamma + c);
    float4 bt = *(const float4*)(beta + c);
    __bf16* yr = y + (size_t)row * DD + c;
    yr[0] = (__bf16)(dx * rstd * g.x + bt.x);
    yr[1] = (__bf16)(dy * rstd * g.y + bt.y);
    yr[2] = (__bf16)(dz * rstd * g.z + bt.z);
    yr[3] = (__bf16)(dw * rstd * g.w + bt.w);
}

// ---------------------------------------------------------------------------
// Weight pre-pass: W f32 [K,N] -> Wt bf16 [N,K] (transpose + convert, once).
// ---------------------------------------------------------------------------
__global__ __launch_bounds__(256) void wconv(const float* __restrict__ W,
                                             __bf16* __restrict__ Wt) {
    __shared__ __bf16 T[64][72];
    const int t  = threadIdx.x;
    const int nb = blockIdx.x * 64;
    const int kb = blockIdx.y * 64;
    #pragma unroll
    for (int it = 0; it < 4; ++it) {
        const int idx = it * 256 + t;       // 0..1023
        const int kl  = idx >> 4;           // 0..63
        const int n4  = (idx & 15) * 4;     // 0..60
        const float4 w = *(const float4*)(W + (size_t)(kb + kl) * DD + nb + n4);
        T[n4 + 0][kl] = (__bf16)w.x;
        T[n4 + 1][kl] = (__bf16)w.y;
        T[n4 + 2][kl] = (__bf16)w.z;
        T[n4 + 3][kl] = (__bf16)w.w;
    }
    __syncthreads();
    const int n = t >> 2, seg = (t & 3) * 16;
    __bf16* dst = Wt + (size_t)(nb + n) * DD + kb + seg;
    *(v8bf*)dst       = *(const v8bf*)&T[n][seg];
    *(v8bf*)(dst + 8) = *(const v8bf*)&T[n][seg + 8];
}

// ---------------------------------------------------------------------------
// GEMM: out = A(bf16 [4096,1024]) @ Wt^T(bf16 stored [N,K]) + bias
//  MODE 0: out f32 [4096,1024];  MODE 1: bf16 -> [B,H,S,HD];  MODE 2: [B,H,HD,S]
// Block tile 64x128, 8 waves (4Mx2N), wave tile 16x64, K-chunk 64,
// double-buffered LDS with async global->LDS staging, hoisted fragments.
// ---------------------------------------------------------------------------
template <int MODE>
__global__ __launch_bounds__(256) void gemm_bf16(const __bf16* __restrict__ A,
                                                 const __bf16* __restrict__ Wt,
                                                 const float*  __restrict__ bias,
                                                 void* __restrict__ out) {
    __shared__ __bf16 As[2][64][72];     // [m][k]
    __shared__ __bf16 Bs[2][128][72];    // [n][k]

    const int t     = threadIdx.x;
    const int lane  = t & 31;
    const int wave  = t >> 5;
    const int waveM = wave & 3;
    const int waveN = wave >> 2;
    const int mBase = blockIdx.y * 64;
    const int nBase = blockIdx.x * 128;
    const int lrow  = lane & 15;
    const int grp   = lane >> 4;

    const __bf16* Abase = A  + (size_t)mBase * DD;
    const __bf16* Bbase = Wt + (size_t)nBase * DD;

    auto stage = [&](int kt, int buf) {
        #pragma unroll
        for (int i = 0; i < 2; ++i) {
            const int seg = i * 256 + t;                 // 0..511
            const int r = seg >> 3, c = (seg & 7) * 8;
            cp16_to_lds(Abase + (size_t)r * DD + kt + c, &As[buf][r][c]);
        }
        #pragma unroll
        for (int i = 0; i < 4; ++i) {
            const int seg = i * 256 + t;                 // 0..1023
            const int r = seg >> 3, c = (seg & 7) * 8;
            cp16_to_lds(Bbase + (size_t)r * DD + kt + c, &Bs[buf][r][c]);
        }
    };

    v8f acc[4];
    #pragma unroll
    for (int i = 0; i < 4; ++i) acc[i] = vzero8();

    stage(0, 0);
    for (int i = 0; i < DD / 64; ++i) {
        wait_stage();
        __syncthreads();
        if (i + 1 < DD / 64) stage((i + 1) * 64, (i + 1) & 1);
        const int buf = i & 1;
        #pragma unroll
        for (int kc = 0; kc < 64; kc += 32) {
            // hoist all fragments of this K-step into distinct registers
            const __bf16* ap = &As[buf][waveM * 16 + lrow][kc + grp * 8];
            v16bf af = join8(*(const v8bf*)ap, *(const v8bf*)(ap + 16));
            v16bf bfr[4];
            #pragma unroll
            for (int nt = 0; nt < 4; ++nt) {
                const __bf16* bp = &Bs[buf][waveN * 64 + nt * 16 + lrow][kc + grp * 8];
                bfr[nt] = join8(*(const v8bf*)bp, *(const v8bf*)(bp + 16));
            }
            #pragma unroll
            for (int nt = 0; nt < 4; ++nt)
                acc[nt] = __builtin_amdgcn_wmma_f32_16x16x32_bf16(
                    false, af, false, bfr[nt], (short)0, acc[nt], false, false);
        }
        __syncthreads();
    }

    const int mgrp = grp * 8;
    #pragma unroll
    for (int nt = 0; nt < 4; ++nt) {
        const int gn = nBase + waveN * 64 + nt * 16 + lrow;
        const float bv = bias[gn];
        #pragma unroll
        for (int r = 0; r < 8; ++r) {
            const int gm = mBase + waveM * 16 + mgrp + r;
            const float val = acc[nt][r] + bv;
            if (MODE == 0) {
                ((float*)out)[(size_t)gm * DD + gn] = val;
            } else {
                const int b = gm >> 11, s = gm & (SS - 1);
                const int h = gn >> 6,  d = gn & (HDD - 1);
                __bf16* o = (__bf16*)out;
                if (MODE == 1)
                    o[(((size_t)(b * HH + h)) * SS + s) * HDD + d] = (__bf16)val;
                else
                    o[(((size_t)(b * HH + h)) * HDD + d) * SS + s] = (__bf16)val;
            }
        }
    }
}

// ---------------------------------------------------------------------------
// Flash attention: 8 waves/block, 128 queries/block, 64-key tiles,
// double-buffered async K/V staging, online softmax, WMMA for QK^T and PV.
// qh/kh: [B*H,S,HD] bf16; vt: [B*H,HD,S] bf16; obuf: [B*H,S,HD] bf16.
// ---------------------------------------------------------------------------
__global__ __launch_bounds__(256) void attn_kernel(const __bf16* __restrict__ qh,
                                                   const __bf16* __restrict__ kh,
                                                   const __bf16* __restrict__ vt,
                                                   const int*    __restrict__ mask,
                                                   __bf16* __restrict__ obuf) {
    __shared__ __bf16 Ks[2][64][72];        // [key][d]
    __shared__ __bf16 Vs[2][64][72];        // [d][key]
    __shared__ __bf16 Ps[8][16][72];        // per-wave probs [q][key]

    const int t    = threadIdx.x;
    const int lane = t & 31;
    const int wave = t >> 5;                // 0..7
    const int bh   = blockIdx.y;            // 0..31
    const int b    = bh >> 4;
    const int qb   = blockIdx.x * 128;
    const int lrow = lane & 15;
    const int grp  = lane >> 4;
    const int mgrp = grp * 8;
    const int qRow0 = qb + wave * 16;

    const __bf16* Kbase = kh + (size_t)bh * SS * HDD;
    const __bf16* Vbase = vt + (size_t)bh * HDD * SS;

    auto stage = [&](int kb, int buf) {
        #pragma unroll
        for (int i = 0; i < 2; ++i) {
            const int seg = i * 256 + t;                 // 0..511
            const int r = seg >> 3, c = (seg & 7) * 8;
            cp16_to_lds(Kbase + (size_t)(kb + r) * HDD + c, &Ks[buf][r][c]);
            cp16_to_lds(Vbase + (size_t)r * SS + kb + c,    &Vs[buf][r][c]);
        }
    };

    // Q fragments (HD=64 -> 2 WMMA K-steps), loaded once from global
    v16bf qf[2];
    {
        const __bf16* qr = qh + ((size_t)bh * SS + qRow0 + lrow) * HDD;
        #pragma unroll
        for (int kc = 0; kc < 2; ++kc)
            qf[kc] = join8(*(const v8bf*)(qr + kc * 32 + grp * 8),
                           *(const v8bf*)(qr + kc * 32 + 16 + grp * 8));
    }

    float rmax[8], rsum[8];
    #pragma unroll
    for (int r = 0; r < 8; ++r) { rmax[r] = -3.0e38f; rsum[r] = 0.f; }
    v8f oacc[4];
    #pragma unroll
    for (int i = 0; i < 4; ++i) oacc[i] = vzero8();

    stage(0, 0);
    for (int ib = 0; ib < SS / 64; ++ib) {
        const int kb = ib * 64, buf = ib & 1;
        wait_stage();
        __syncthreads();
        if (ib + 1 < SS / 64) stage(kb + 64, (ib + 1) & 1);

        // S = Q @ K^T  (contraction over HD), hoisted fragments per K-step
        v8f sc[4];
        #pragma unroll
        for (int i = 0; i < 4; ++i) sc[i] = vzero8();
        #pragma unroll
        for (int kc = 0; kc < 2; ++kc) {
            v16bf bfr[4];
            #pragma unroll
            for (int nt = 0; nt < 4; ++nt) {
                const __bf16* bp = &Ks[buf][nt * 16 + lrow][kc * 32 + grp * 8];
                bfr[nt] = join8(*(const v8bf*)bp, *(const v8bf*)(bp + 16));
            }
            #pragma unroll
            for (int nt = 0; nt < 4; ++nt)
                sc[nt] = __builtin_amdgcn_wmma_f32_16x16x32_bf16(
                    false, qf[kc], false, bfr[nt], (short)0, sc[nt], false, false);
        }

        // mask (before scale, per reference) + online softmax
        float p[4][8];
        #pragma unroll
        for (int nt = 0; nt < 4; ++nt) {
            const int key = kb + nt * 16 + lrow;
            const int* mcol = mask + (size_t)b * SS * SS + key;
            #pragma unroll
            for (int r = 0; r < 8; ++r) {
                const int q = qRow0 + mgrp + r;
                const int m = mcol[(size_t)q * SS];
                p[nt][r] = (m == 0 ? NEGV : sc[nt][r]) * SCALE;
            }
        }
        #pragma unroll
        for (int r = 0; r < 8; ++r) {
            float mx = fmaxf(fmaxf(p[0][r], p[1][r]), fmaxf(p[2][r], p[3][r]));
            #pragma unroll
            for (int off = 8; off >= 1; off >>= 1) mx = fmaxf(mx, __shfl_xor(mx, off));
            const float nm = fmaxf(rmax[r], mx);
            const float corr = __expf(rmax[r] - nm);
            rmax[r] = nm;
            float sum = 0.f;
            #pragma unroll
            for (int nt = 0; nt < 4; ++nt) {
                const float e = __expf(p[nt][r] - nm);
                p[nt][r] = e;
                sum += e;
            }
            #pragma unroll
            for (int off = 8; off >= 1; off >>= 1) sum += __shfl_xor(sum, off);
            rsum[r] = rsum[r] * corr + sum;
            #pragma unroll
            for (int nt = 0; nt < 4; ++nt) oacc[nt][r] *= corr;
        }

        // C-layout -> A-layout transpose of P via per-wave LDS (wave-local fence)
        #pragma unroll
        for (int nt = 0; nt < 4; ++nt)
            #pragma unroll
            for (int r = 0; r < 8; ++r)
                Ps[wave][mgrp + r][nt * 16 + lrow] = (__bf16)p[nt][r];
        wave_lds_fence();

        // O += P @ V (contraction over keys), hoisted fragments per K-step
        #pragma unroll
        for (int kc = 0; kc < 2; ++kc) {
            const __bf16* ap = &Ps[wave][lrow][kc * 32 + grp * 8];
            v16bf af = join8(*(const v8bf*)ap, *(const v8bf*)(ap + 16));
            v16bf bfr[4];
            #pragma unroll
            for (int nt = 0; nt < 4; ++nt) {
                const __bf16* bp = &Vs[buf][nt * 16 + lrow][kc * 32 + grp * 8];
                bfr[nt] = join8(*(const v8bf*)bp, *(const v8bf*)(bp + 16));
            }
            #pragma unroll
            for (int nt = 0; nt < 4; ++nt)
                oacc[nt] = __builtin_amdgcn_wmma_f32_16x16x32_bf16(
                    false, af, false, bfr[nt], (short)0, oacc[nt], false, false);
        }
    }

    // normalize; store O in [B,H,S,HD] contiguous (raw reshape feeds Wo gemm)
    #pragma unroll
    for (int nt = 0; nt < 4; ++nt) {
        const int d = nt * 16 + lrow;
        #pragma unroll
        for (int r = 0; r < 8; ++r) {
            const int q = qRow0 + mgrp + r;
            obuf[((size_t)bh * SS + q) * HDD + d] = (__bf16)(oacc[nt][r] / rsum[r]);
        }
    }
}

// ---------------------------------------------------------------------------
extern "C" void kernel_launch(void* const* d_in, const int* in_sizes, int n_in,
                              void* d_out, int out_size, void* d_ws, size_t ws_size,
                              hipStream_t stream) {
    const float* q    = (const float*)d_in[0];
    const float* k    = (const float*)d_in[1];
    const float* v    = (const float*)d_in[2];
    const int*   mask = (const int*)  d_in[3];
    const float* Wq   = (const float*)d_in[4];
    const float* bq   = (const float*)d_in[5];
    const float* Wk   = (const float*)d_in[6];
    const float* bk   = (const float*)d_in[7];
    const float* Wv   = (const float*)d_in[8];
    const float* bv   = (const float*)d_in[9];
    const float* Wo   = (const float*)d_in[10];
    const float* bo   = (const float*)d_in[11];
    const float* gamma= (const float*)d_in[12];
    const float* beta = (const float*)d_in[13];

    char* ws = (char*)d_ws;
    const size_t BUF  = (size_t)ROWS * DD * sizeof(__bf16);  // 8 MB
    const size_t WBUF = (size_t)DD * DD * sizeof(__bf16);    // 2 MB
    __bf16* xq   = (__bf16*)(ws + 0 * BUF);
    __bf16* xk   = (__bf16*)(ws + 1 * BUF);
    __bf16* xv   = (__bf16*)(ws + 2 * BUF);
    __bf16* qh   = (__bf16*)(ws + 3 * BUF);
    __bf16* kh   = (__bf16*)(ws + 4 * BUF);
    __bf16* vt   = (__bf16*)(ws + 5 * BUF);
    __bf16* obuf = (__bf16*)(ws + 6 * BUF);
    __bf16* WqT  = (__bf16*)(ws + 7 * BUF);
    __bf16* WkT  = (__bf16*)(ws + 7 * BUF + 1 * WBUF);
    __bf16* WvT  = (__bf16*)(ws + 7 * BUF + 2 * WBUF);
    __bf16* WoT  = (__bf16*)(ws + 7 * BUF + 3 * WBUF);

    const dim3 gLN(ROWS), bLN(256);
    const dim3 gWC(DD / 64, DD / 64), bWC(256);
    const dim3 gGE(DD / 128, ROWS / 64), bGE(256);
    const dim3 gAT(SS / 128, BB * HH), bAT(256);

    wconv<<<gWC, bWC, 0, stream>>>(Wq, WqT);
    wconv<<<gWC, bWC, 0, stream>>>(Wk, WkT);
    wconv<<<gWC, bWC, 0, stream>>>(Wv, WvT);
    wconv<<<gWC, bWC, 0, stream>>>(Wo, WoT);

    ln_bf16<<<gLN, bLN, 0, stream>>>(q, gamma, beta, xq);
    ln_bf16<<<gLN, bLN, 0, stream>>>(k, gamma, beta, xk);
    ln_bf16<<<gLN, bLN, 0, stream>>>(v, gamma, beta, xv);

    gemm_bf16<1><<<gGE, bGE, 0, stream>>>(xq, WqT, bq, (void*)qh);
    gemm_bf16<1><<<gGE, bGE, 0, stream>>>(xk, WkT, bk, (void*)kh);
    gemm_bf16<2><<<gGE, bGE, 0, stream>>>(xv, WvT, bv, (void*)vt);

    attn_kernel<<<gAT, bAT, 0, stream>>>(qh, kh, vt, mask, obuf);

    gemm_bf16<0><<<gGE, bGE, 0, stream>>>(obuf, WoT, bo, d_out);
}
